// MultiHeadAttention_69698729280159
// MI455X (gfx1250) — matrix operand order repaired and checked by reading, not compiled, not run on hardware
//
#include <hip/hip_runtime.h>

// ---------------------------------------------------------------------------
// MultiHeadAttention for MI455X (gfx1250, wave32, WMMA bf16 16x16x32)
// B=2, S=2048, D=1024, H=16, HD=64
// v4 == v2 structure: 32-row tiles per wave (2x register-level reuse of
// B/K/V fragments), NO manual fragment double-buffering (v3 showed it
// spills to scratch; 8-waves/SIMD occupancy is the latency hider).
// ---------------------------------------------------------------------------

typedef __attribute__((ext_vector_type(16))) __bf16 v16bf;
typedef __attribute__((ext_vector_type(8)))  __bf16 v8bf;
typedef __attribute__((ext_vector_type(8)))  float  v8f;

#define Bc 2
#define Sc 2048
#define Dc 1024
#define Hc 16
#define HDc 64

__device__ inline v8bf ld8(const __bf16* p) { return *(const v8bf*)p; }

__device__ inline v16bf mk16(v8bf lo, v8bf hi) {
    union { v16bf v; v8bf h[2]; } u;
    u.h[0] = lo; u.h[1] = hi;
    return u.v;
}

__device__ inline v8f wmma_bf16(v16bf a, v16bf b, v8f c) {
    // D = A(16x32 bf16) * B(32x16 bf16) + C(16x16 f32)
    return __builtin_amdgcn_wmma_f32_16x16x32_bf16(
        false, a, false, b, (short)0, c, false, false);
}

__device__ inline float rmax16(float v) {
    v = fmaxf(v, __shfl_xor(v, 1, 32));
    v = fmaxf(v, __shfl_xor(v, 2, 32));
    v = fmaxf(v, __shfl_xor(v, 4, 32));
    v = fmaxf(v, __shfl_xor(v, 8, 32));
    return v;
}
__device__ inline float rsum16(float v) {
    v += __shfl_xor(v, 1, 32);
    v += __shfl_xor(v, 2, 32);
    v += __shfl_xor(v, 4, 32);
    v += __shfl_xor(v, 8, 32);
    return v;
}

// ------------------------- precision-convert kernels -----------------------

__global__ void cvt_x_kernel(const float* __restrict__ in, __bf16* __restrict__ out, int n) {
    int i = blockIdx.x * 256 + threadIdx.x;
    if (i < n) out[i] = (__bf16)in[i];
}

// W[k][n] fp32 -> WT[n][k] bf16  (coalesced writes; one-time cost, L2-resident)
__global__ void wt_cvt_kernel(const float* __restrict__ W, __bf16* __restrict__ WT) {
    int i = blockIdx.x * 256 + threadIdx.x;   // i = n*1024 + k
    int n = i >> 10, k = i & 1023;
    WT[i] = (__bf16)W[k * Dc + n];
}

// ------------------------- wave GEMM core (32x64 tile) ---------------------
// arow0/arow1: A row base for lane row m of m-subtile 0/1 (bf16, ld=1024)
// bbase: WT + (nquad*64)*1024  (WT is [n][k] so B-fragment K-dim is contiguous)
// c[0..3] = subtile0 x ntiles 0..3 ; c[4..7] = subtile1 x ntiles 0..3
__device__ inline void gemm_32x64(const __bf16* __restrict__ arow0,
                                  const __bf16* __restrict__ arow1,
                                  const __bf16* __restrict__ bbase,
                                  v8f c[8], int lane) {
    const int hh = lane >> 4;
    const int n  = lane & 15;
    const __bf16* br0 = bbase + (size_t)(n +  0) * Dc + 16 * hh;
    const __bf16* br1 = bbase + (size_t)(n + 16) * Dc + 16 * hh;
    const __bf16* br2 = bbase + (size_t)(n + 32) * Dc + 16 * hh;
    const __bf16* br3 = bbase + (size_t)(n + 48) * Dc + 16 * hh;
    for (int kc = 0; kc < Dc; kc += 32) {
        v16bf a0 = mk16(ld8(arow0 + kc + 16 * hh), ld8(arow0 + kc + 8 + 16 * hh));
        v16bf a1 = mk16(ld8(arow1 + kc + 16 * hh), ld8(arow1 + kc + 8 + 16 * hh));
        v16bf b0 = mk16(ld8(br0 + kc), ld8(br0 + kc + 8));
        c[0] = wmma_bf16(a0, b0, c[0]);
        c[4] = wmma_bf16(a1, b0, c[4]);
        v16bf b1 = mk16(ld8(br1 + kc), ld8(br1 + kc + 8));
        c[1] = wmma_bf16(a0, b1, c[1]);
        c[5] = wmma_bf16(a1, b1, c[5]);
        v16bf b2 = mk16(ld8(br2 + kc), ld8(br2 + kc + 8));
        c[2] = wmma_bf16(a0, b2, c[2]);
        c[6] = wmma_bf16(a1, b2, c[6]);
        v16bf b3 = mk16(ld8(br3 + kc), ld8(br3 + kc + 8));
        c[3] = wmma_bf16(a0, b3, c[3]);
        c[7] = wmma_bf16(a1, b3, c[7]);
    }
}

// ------------------------- QKV projection ----------------------------------
// grid: (16, 16, 3)  block: 256 (8 waves).  Each wave: one 32(M) x 64(N) tile.
// N-quad == one head (64 cols). Q,K -> [B,H,S,HD] bf16 ; V -> [B,H,HD,S] bf16.
__global__ void qkv_gemm_kernel(const __bf16* __restrict__ xb,
                                const __bf16* __restrict__ WqT,
                                const __bf16* __restrict__ WkT,
                                const __bf16* __restrict__ WvT,
                                const float* __restrict__ bq,
                                const float* __restrict__ bk,
                                const float* __restrict__ bv,
                                __bf16* __restrict__ Qb,
                                __bf16* __restrict__ Kb,
                                __bf16* __restrict__ Vt) {
    const int lane = threadIdx.x & 31;
    const int wid  = threadIdx.x >> 5;
    const int mt   = blockIdx.x * 8 + wid;   // 0..127  m-tile of 32 rows
    const int nq   = blockIdx.y;             // 0..15   head index
    const int z    = blockIdx.z;             // 0=Q 1=K 2=V

    const __bf16* WT   = (z == 0) ? WqT : (z == 1) ? WkT : WvT;
    const float*  bias = (z == 0) ? bq  : (z == 1) ? bk  : bv;

    const int m  = lane & 15;
    const int hh = lane >> 4;

    v8f c[8] = {v8f{}, v8f{}, v8f{}, v8f{}, v8f{}, v8f{}, v8f{}, v8f{}};
    gemm_32x64(xb + ((size_t)mt * 32 + m) * Dc,
               xb + ((size_t)mt * 32 + 16 + m) * Dc,
               WT + (size_t)nq * 64 * Dc, c, lane);

    float bv4[4];
#pragma unroll
    for (int nt = 0; nt < 4; nt++) bv4[nt] = bias[nq * 64 + nt * 16 + m];

    const int b = (mt * 32) >> 11;  // tiles never straddle batches (2048%32==0)
#pragma unroll
    for (int mh = 0; mh < 2; mh++) {
#pragma unroll
        for (int r = 0; r < 8; r++) {
            const int row = mt * 32 + mh * 16 + r + 8 * hh;
            const int s   = row & (Sc - 1);
#pragma unroll
            for (int nt = 0; nt < 4; nt++) {
                const int d = nt * 16 + m;
                const float val = c[mh * 4 + nt][r] + bv4[nt];
                if (z == 0)
                    Qb[(((size_t)(b * Hc + nq)) * Sc + s) * HDc + d] = (__bf16)val;
                else if (z == 1)
                    Kb[(((size_t)(b * Hc + nq)) * Sc + s) * HDc + d] = (__bf16)val;
                else
                    Vt[(((size_t)(b * Hc + nq)) * HDc + d) * Sc + s] = (__bf16)val;
            }
        }
    }
}

// ------------------------- flash attention ---------------------------------
// One wave per (b,h, 32-row q-tile). 2048 tiles total. 256 blocks x 8 waves.
// K/V fragments are reused across the two 16-row q-subtiles (2x reuse).
// Online softmax (fp32), causal mask computed analytically.
#define PLDS_STRIDE 40   // bf16 elements per row (16B-aligned rows, de-conflicted)
__global__ void flash_attn_kernel(const __bf16* __restrict__ Qb,
                                  const __bf16* __restrict__ Kb,
                                  const __bf16* __restrict__ Vt,
                                  __bf16* __restrict__ Ob) {
    __shared__ __bf16 lds[8 * 32 * PLDS_STRIDE];

    const int lane = threadIdx.x & 31;
    const int wid  = threadIdx.x >> 5;
    const int t    = blockIdx.x * 8 + wid;   // 0..2047
    const int qt   = t & 63;                 // 32-row q tile within (b,h)
    const int bh   = t >> 6;                 // b*H + h

    const int m  = lane & 15;
    const int hh = lane >> 4;

    // Q tile: 2 m-subtiles x 2 A-fragments covering head-dim 0..63
    const __bf16* qbase = Qb + (((size_t)bh) * Sc + qt * 32) * HDc;
    v16bf qa[2][2];
#pragma unroll
    for (int mt = 0; mt < 2; mt++) {
        const __bf16* p = qbase + (size_t)(mt * 16 + m) * HDc;
        qa[mt][0] = mk16(ld8(p +  0 + 16 * hh), ld8(p +  8 + 16 * hh));
        qa[mt][1] = mk16(ld8(p + 32 + 16 * hh), ld8(p + 40 + 16 * hh));
    }

    v8f o[2][4];
    float mrow[2][8], lrow[2][8];
#pragma unroll
    for (int mt = 0; mt < 2; mt++) {
#pragma unroll
        for (int dt = 0; dt < 4; dt++) o[mt][dt] = v8f{};
#pragma unroll
        for (int r = 0; r < 8; r++) { mrow[mt][r] = -1e30f; lrow[mt][r] = 0.0f; }
    }

    __bf16* pl = lds + wid * 32 * PLDS_STRIDE;
    const __bf16* kbh = Kb + ((size_t)bh) * Sc * HDc;
    const __bf16* vbh = Vt + ((size_t)bh) * HDc * Sc;

    const int kend = qt * 32 + 32;  // number of allowed key positions

    for (int c = 0; c < kend; c += 32) {
        // ---- K fragments for cols c..c+31 (shared by both q-subtiles) ----
        v16bf kf[2][2];
#pragma unroll
        for (int kt = 0; kt < 2; kt++) {
            const __bf16* p = kbh + (size_t)(c + kt * 16 + m) * HDc + 16 * hh;
            kf[kt][0] = mk16(ld8(p +  0), ld8(p +  8));
            kf[kt][1] = mk16(ld8(p + 32), ld8(p + 40));
        }
        // speculative prefetch of next chunk (gfx1250 global_prefetch path)
        __builtin_prefetch(kbh + (size_t)(c + 32 + m) * HDc + 16 * hh, 0, 1);
        __builtin_prefetch(vbh + (size_t)m * Sc + c + 32 + 16 * hh, 0, 1);

        // ---- scores: 2 m-subtiles x 2 k-subtiles ----
        v8f s[2][2];
#pragma unroll
        for (int mt = 0; mt < 2; mt++) {
#pragma unroll
            for (int kt = 0; kt < 2; kt++) {
                v8f acc = v8f{};
                acc = wmma_bf16(qa[mt][0], kf[kt][0], acc);
                acc = wmma_bf16(qa[mt][1], kf[kt][1], acc);
                s[mt][kt] = acc;
            }
        }

        // ---- scale + causal mask, online softmax, P -> LDS ----
        const int kk = c + m;
#pragma unroll
        for (int mt = 0; mt < 2; mt++) {
            const int qrow_base = qt * 32 + mt * 16 + 8 * hh;
#pragma unroll
            for (int r = 0; r < 8; r++) {
                const int q = qrow_base + r;
                s[mt][0][r] = (kk      <= q) ? s[mt][0][r] * 0.125f : -1e30f;
                s[mt][1][r] = (kk + 16 <= q) ? s[mt][1][r] * 0.125f : -1e30f;
            }
#pragma unroll
            for (int r = 0; r < 8; r++) {
                float mx    = rmax16(fmaxf(s[mt][0][r], s[mt][1][r]));
                float mnew  = fmaxf(mrow[mt][r], mx);
                float alpha = __expf(mrow[mt][r] - mnew);
                float p0 = __expf(s[mt][0][r] - mnew);
                float p1 = __expf(s[mt][1][r] - mnew);
                float ps = rsum16(p0 + p1);
                lrow[mt][r] = lrow[mt][r] * alpha + ps;
                mrow[mt][r] = mnew;
                o[mt][0][r] *= alpha; o[mt][1][r] *= alpha;
                o[mt][2][r] *= alpha; o[mt][3][r] *= alpha;
                pl[(mt * 16 + r + 8 * hh) * PLDS_STRIDE +      m] = (__bf16)p0;
                pl[(mt * 16 + r + 8 * hh) * PLDS_STRIDE + 16 + m] = (__bf16)p1;
            }
        }

        // ---- reshape P (C/D layout) -> A-fragment layout via LDS ----
        v16bf pa[2];
#pragma unroll
        for (int mt = 0; mt < 2; mt++) {
            const __bf16* pr = pl + (size_t)(mt * 16 + m) * PLDS_STRIDE;
            pa[mt] = mk16(ld8(pr + 16 * hh), ld8(pr + 8 + 16 * hh));
        }

        // ---- O += P(16x32) * V(32x64), V fragments shared across subtiles --
#pragma unroll
        for (int dt = 0; dt < 4; dt++) {
            const __bf16* p = vbh + (size_t)(dt * 16 + m) * Sc + c + 16 * hh;
            v16bf vf = mk16(ld8(p), ld8(p + 8));
            o[0][dt] = wmma_bf16(pa[0], vf, o[0][dt]);
            o[1][dt] = wmma_bf16(pa[1], vf, o[1][dt]);
        }
    }

    // ---- epilogue: normalize, merge heads into [B,S,H*HD] bf16 ----
    const int b = bh >> 4, h = bh & 15;
#pragma unroll
    for (int mt = 0; mt < 2; mt++) {
#pragma unroll
        for (int r = 0; r < 8; r++) {
            const float inv = 1.0f / lrow[mt][r];
            const int   s   = qt * 32 + mt * 16 + r + 8 * hh;
            const size_t base = ((size_t)b * Sc + s) * Dc + h * HDc;
#pragma unroll
            for (int dt = 0; dt < 4; dt++)
                Ob[base + dt * 16 + m] = (__bf16)(o[mt][dt][r] * inv);
        }
    }
}

// ------------------------- output projection -------------------------------
__global__ void out_proj_kernel(const __bf16* __restrict__ Ob,
                                const __bf16* __restrict__ WoT,
                                const float* __restrict__ bo,
                                float* __restrict__ out) {
    const int lane = threadIdx.x & 31;
    const int wid  = threadIdx.x >> 5;
    const int mt   = blockIdx.x * 8 + wid;   // 0..127
    const int nq   = blockIdx.y;
    const int m  = lane & 15;
    const int hh = lane >> 4;

    v8f c[8] = {v8f{}, v8f{}, v8f{}, v8f{}, v8f{}, v8f{}, v8f{}, v8f{}};
    gemm_32x64(Ob + ((size_t)mt * 32 + m) * Dc,
               Ob + ((size_t)mt * 32 + 16 + m) * Dc,
               WoT + (size_t)nq * 64 * Dc, c, lane);

    float bv4[4];
#pragma unroll
    for (int nt = 0; nt < 4; nt++) bv4[nt] = bo[nq * 64 + nt * 16 + m];

#pragma unroll
    for (int mh = 0; mh < 2; mh++) {
#pragma unroll
        for (int r = 0; r < 8; r++) {
            const int row = mt * 32 + mh * 16 + r + 8 * hh;
#pragma unroll
            for (int nt = 0; nt < 4; nt++)
                out[(size_t)row * Dc + nq * 64 + nt * 16 + m] =
                    c[mh * 4 + nt][r] + bv4[nt];
        }
    }
}

// ---------------------------------------------------------------------------

extern "C" void kernel_launch(void* const* d_in, const int* in_sizes, int n_in,
                              void* d_out, int out_size, void* d_ws, size_t ws_size,
                              hipStream_t stream) {
    const float* x  = (const float*)d_in[0];
    // d_in[1] = attn_mask (causal, computed analytically in-kernel)
    const float* Wq = (const float*)d_in[2];
    const float* bq = (const float*)d_in[3];
    const float* Wk = (const float*)d_in[4];
    const float* bk = (const float*)d_in[5];
    const float* Wv = (const float*)d_in[6];
    const float* bv = (const float*)d_in[7];
    const float* Wo = (const float*)d_in[8];
    const float* bo = (const float*)d_in[9];
    float* out = (float*)d_out;

    char* ws = (char*)d_ws;
    const size_t MB = 1u << 20;
    __bf16* xb  = (__bf16*)(ws +  0 * MB);   // [4096][1024]           8 MB
    __bf16* WqT = (__bf16*)(ws +  8 * MB);   // [1024][1024] (n-major) 2 MB
    __bf16* WkT = (__bf16*)(ws + 10 * MB);
    __bf16* WvT = (__bf16*)(ws + 12 * MB);
    __bf16* WoT = (__bf16*)(ws + 14 * MB);
    __bf16* Qb  = (__bf16*)(ws + 16 * MB);   // [B,H,S,HD]             8 MB
    __bf16* Kb  = (__bf16*)(ws + 24 * MB);   // [B,H,S,HD]             8 MB
    __bf16* Vt  = (__bf16*)(ws + 32 * MB);   // [B,H,HD,S]             8 MB
    __bf16* Ob  = (__bf16*)(ws + 40 * MB);   // [B,S,D]                8 MB

    const int NX = Bc * Sc * Dc;  // 4194304
    cvt_x_kernel<<<NX / 256, 256, 0, stream>>>(x, xb, NX);

    const int NW = Dc * Dc;       // 1048576
    wt_cvt_kernel<<<NW / 256, 256, 0, stream>>>(Wq, WqT);
    wt_cvt_kernel<<<NW / 256, 256, 0, stream>>>(Wk, WkT);
    wt_cvt_kernel<<<NW / 256, 256, 0, stream>>>(Wv, WvT);
    wt_cvt_kernel<<<NW / 256, 256, 0, stream>>>(Wo, WoT);

    qkv_gemm_kernel<<<dim3(16, 16, 3), 256, 0, stream>>>(
        xb, WqT, WkT, WvT, bq, bk, bv, Qb, Kb, Vt);

    flash_attn_kernel<<<256, 256, 0, stream>>>(Qb, Kb, Vt, Ob);

    out_proj_kernel<<<dim3(16, 16, 1), 256, 0, stream>>>(Ob, WoT, bo, out);
}
